// Transformer_88021059764228
// MI455X (gfx1250) — matrix operand order repaired and checked by reading, not compiled, not run on hardware
//
#include <hip/hip_runtime.h>
#include <stdint.h>

// ---------------------------------------------------------------------------
// CDNA5 (gfx1250) Transformer-XL forward, bf16 WMMA (V_WMMA_F32_16X16X32_BF16)
// wave32; each wave computes a 16x(16*NB) C-strip: A fragment reused across
// NB WMMAs (raises FLOP/byte from L2, the binding resource since the whole
// ~145MB working set is resident in the 192MB L2).
// ---------------------------------------------------------------------------

typedef __attribute__((ext_vector_type(16))) __bf16       v16bf;
typedef __attribute__((ext_vector_type(8)))  float        v8f;
typedef __attribute__((ext_vector_type(4)))  unsigned int u32x4;

#define S_LEN   507      // pooled sequence length
#define SPAD    512      // padded S for 16-tiles
#define BSZ     16
#define DMODEL  128
#define NHEAD   4
#define DHEAD   32
#define DINNER  512
#define RLEN    1013     // 2*S-1
#define RPAD    1024
#define MROWS   (S_LEN * BSZ)   // 8112, multiple of 16

__device__ __forceinline__ unsigned short f2bf(float f) {
    union { float f; unsigned u; } v; v.f = f;
    unsigned u = v.u;
    return (unsigned short)((u + 0x7FFFu + ((u >> 16) & 1u)) >> 16);  // RNE
}

union Frag { v16bf v; u32x4 q[2]; };

// ---------------------------------------------------------------------------
// Generic bf16 WMMA GEMM:  C = A[M,K] * Bt[N,K]^T  (+bias +residual, relu)
// A, Bt row-major bf16 (ushort). Batch offsets use (batch>>2, batch&3) split
// so the attention layouts (b,n) map with two strides. NB = N-tiles per wave.
// ---------------------------------------------------------------------------
template <int NB>
__global__ void __launch_bounds__(128)
gemm_wmma(const unsigned short* __restrict__ A, int lda, long long aO, long long aI,
          const unsigned short* __restrict__ Bt, int ldb, long long bO, long long bI,
          float* __restrict__ outf, unsigned short* __restrict__ outb,
          int ldc, long long cO, long long cI,
          const float* __restrict__ bias, const float* __restrict__ res,
          int M, int N, int K, int Mvalid, int relu)
{
    int batch = blockIdx.y;
    A  += (long long)(batch >> 2) * aO + (long long)(batch & 3) * aI;
    Bt += (long long)(batch >> 2) * bO + (long long)(batch & 3) * bI;
    long long cBase = (long long)(batch >> 2) * cO + (long long)(batch & 3) * cI;

    int tilesNb = (N >> 4) / NB;                    // N-tile groups
    int tile = blockIdx.x * (blockDim.x >> 5) + (threadIdx.x >> 5);
    if (tile >= (M >> 4) * tilesNb) return;         // wave-uniform
    int m0 = (tile / tilesNb) << 4;
    int n0 = (tile % tilesNb) * (16 * NB);

    int lane = threadIdx.x & 31;
    int grp  = lane >> 4;       // K-group select
    int col  = lane & 15;

    // A fragment: row = m0+col; k = kk + grp*16 + (0..15) contiguous
    const unsigned short* arow = A  + (long long)(m0 + col) * lda + grp * 16;
    // B fragment: Bt row = n0 + 16*t + col; k = kk + grp*16 + (0..15) contiguous
    const unsigned short* brow = Bt + (long long)(n0 + col) * ldb + grp * 16;

    const v8f vzero = {0.f, 0.f, 0.f, 0.f, 0.f, 0.f, 0.f, 0.f};
    v8f acc[NB];
    #pragma unroll
    for (int t = 0; t < NB; ++t) acc[t] = vzero;

    for (int kk = 0; kk < K; kk += 32) {
        if (kk + 32 < K) {                          // prefetch next K chunk
            __builtin_prefetch(arow + kk + 32, 0, 3);
            __builtin_prefetch(brow + kk + 32, 0, 3);
        }
        Frag a;
        a.q[0] = *reinterpret_cast<const u32x4*>(arow + kk);
        a.q[1] = *reinterpret_cast<const u32x4*>(arow + kk + 8);
        #pragma unroll
        for (int t = 0; t < NB; ++t) {
            Frag b;
            const u32x4* bp = reinterpret_cast<const u32x4*>(
                brow + (long long)(16 * t) * ldb + kk);
            b.q[0] = bp[0];
            b.q[1] = bp[1];
            acc[t] = __builtin_amdgcn_wmma_f32_16x16x32_bf16(
                false, a.v, false, b.v, (short)0, acc[t], false, false);
        }
    }

    #pragma unroll
    for (int t = 0; t < NB; ++t) {
        int n = n0 + 16 * t + col;
        float bv = bias ? bias[n] : 0.0f;
        #pragma unroll
        for (int v = 0; v < 8; ++v) {
            int m = m0 + grp * 8 + v;
            if (m >= Mvalid) continue;
            float val = acc[t][v] + bv;
            long long idx = cBase + (long long)m * ldc + n;
            if (res)  val += res[idx];
            if (relu) val = fmaxf(val, 0.0f);
            if (outf) outf[idx] = val;
            if (outb) outb[idx] = f2bf(val);
        }
    }
}

// ---------------------------------------------------------------------------
// BD term with fused Transformer-XL rel_shift:
//   scores[bn][i][j] += sum_d Qbd[bn][i][d] * rk[r][n*32+d],  r = j - i + S-1
// Computed as raw (i,r) tiles (4 r-tiles per wave, A reused), scatter-added
// at j = r + i - (S-1). Each (i,r) maps to a unique (i,j): no atomics needed.
// ---------------------------------------------------------------------------
__global__ void __launch_bounds__(128)
bd_wmma(const unsigned short* __restrict__ Qbd, const unsigned short* __restrict__ rkb,
        float* __restrict__ scores)
{
    const int NB = 4;
    int bn = blockIdx.y;
    const unsigned short* A  = Qbd + (long long)bn * (SPAD * DHEAD);
    const unsigned short* Bt = rkb + (bn & 3) * DHEAD;   // head slice of [RPAD,128]
    float* sc = scores + (long long)bn * (SPAD * SPAD);

    const int tilesNb = (RPAD >> 4) / NB;   // 16
    int tile = blockIdx.x * (blockDim.x >> 5) + (threadIdx.x >> 5);
    if (tile >= (SPAD >> 4) * tilesNb) return;
    int m0 = (tile / tilesNb) << 4;
    int n0 = (tile % tilesNb) * (16 * NB);

    int lane = threadIdx.x & 31;
    int grp  = lane >> 4;
    int col  = lane & 15;

    const unsigned short* arow = A  + (m0 + col) * DHEAD  + grp * 16;
    const unsigned short* brow = Bt + (long long)(n0 + col) * DMODEL + grp * 16;

    Frag a;
    a.q[0] = *reinterpret_cast<const u32x4*>(arow);
    a.q[1] = *reinterpret_cast<const u32x4*>(arow + 8);

    const v8f vzero = {0.f, 0.f, 0.f, 0.f, 0.f, 0.f, 0.f, 0.f};
    v8f acc[NB];
    #pragma unroll
    for (int t = 0; t < NB; ++t) {
        Frag b;
        const u32x4* bp = reinterpret_cast<const u32x4*>(
            brow + (long long)(16 * t) * DMODEL);
        b.q[0] = bp[0];
        b.q[1] = bp[1];
        acc[t] = __builtin_amdgcn_wmma_f32_16x16x32_bf16(
            false, a.v, false, b.v, (short)0, vzero, false, false);
    }

    #pragma unroll
    for (int t = 0; t < NB; ++t) {
        int r = n0 + 16 * t + col;
        #pragma unroll
        for (int v = 0; v < 8; ++v) {
            int i = m0 + grp * 8 + v;
            int j = r + i - (S_LEN - 1);
            if (i < S_LEN && j >= 0 && j < S_LEN) {
                long long idx = (long long)i * SPAD + j;
                sc[idx] += acc[t][v];
            }
        }
    }
}

// ---------------------------------------------------------------------------
// Conv1d(K=11) + bias + ReLU + AvgPool(2) -> w[(s*16+b)][c] (f32 + bf16)
// ---------------------------------------------------------------------------
__global__ void conv_front(const float* __restrict__ x, const float* __restrict__ cw,
                           const float* __restrict__ cb, float* __restrict__ wf,
                           unsigned short* __restrict__ wb)
{
    int t = blockIdx.x * blockDim.x + threadIdx.x;
    if (t >= MROWS * DMODEL) return;
    int c  = t & 127;
    int sb = t >> 7;
    int b  = sb & 15;
    int s  = sb >> 4;
    const float* xr = x + b * 1024;
    const float* kr = cw + c * 11;
    float bias = cb[c];
    float h0 = bias, h1 = bias;
    int l0 = 2 * s;
    #pragma unroll
    for (int k = 0; k < 11; ++k) {
        float wv = kr[k];
        h0 += xr[l0 + k] * wv;
        h1 += xr[l0 + 1 + k] * wv;
    }
    float val = 0.5f * (fmaxf(h0, 0.f) + fmaxf(h1, 0.f));
    wf[t] = val;
    wb[t] = f2bf(val);
}

// Relative positional embedding, padded to [RPAD,128], bf16.
// inv_freq = 10000^{-(2j)} (raw arange exponent, as in the reference).
__global__ void posemb_kernel(unsigned short* __restrict__ posb)
{
    int t = blockIdx.x * blockDim.x + threadIdx.x;
    if (t >= RPAD * DMODEL) return;
    int r = t >> 7, c = t & 127;
    float val = 0.f;
    if (r < RLEN) {
        float pos = (float)(S_LEN - 1 - r);       // pos_seq[r]
        int j = c & 63;
        float invf = powf(10000.0f, -2.0f * (float)j);
        float a = pos * invf;
        val = (c < 64) ? sinf(a) : cosf(a);
    }
    posb[t] = f2bf(val);
}

// f32 [R,C] -> bf16 [C,R]  (weight pre-transpose for contiguous B fragments)
__global__ void transpose_f32_bf16(const float* __restrict__ in,
                                   unsigned short* __restrict__ out, int R, int C)
{
    int t = blockIdx.x * blockDim.x + threadIdx.x;
    if (t >= R * C) return;
    int r = t / C, c = t % C;
    out[(long long)c * R + r] = f2bf(in[t]);
}

// Split qkv [8112,384] into padded per-(b,n) bf16 operands (+ rel biases).
__global__ void qkv_prep(const float* __restrict__ qkvf, const float* __restrict__ rwb,
                         const float* __restrict__ rrb,
                         unsigned short* __restrict__ Qac, unsigned short* __restrict__ Qbd,
                         unsigned short* __restrict__ Kt, unsigned short* __restrict__ Vt)
{
    int t = blockIdx.x * blockDim.x + threadIdx.x;
    if (t >= 64 * SPAD * DHEAD) return;
    int d  = t & 31;
    int s  = (t >> 5) & 511;
    int bn = t >> 14;
    int b = bn >> 2, n = bn & 3;
    float v = 0.f;
    unsigned short qa = 0, qb = 0, kk = 0;
    if (s < S_LEN) {
        long long base = (long long)(s * BSZ + b) * 384 + n * DHEAD + d;
        float q = qkvf[base];
        float k = qkvf[base + 128];
        v = qkvf[base + 256];
        qa = f2bf(q + rwb[n * DHEAD + d]);
        qb = f2bf(q + rrb[n * DHEAD + d]);
        kk = f2bf(k);
    }
    Qac[t] = qa; Qbd[t] = qb; Kt[t] = kk;
    Vt[(long long)bn * (DHEAD * SPAD) + d * SPAD + s] = (s < S_LEN) ? f2bf(v)
                                                                    : (unsigned short)0;
}

// Row softmax over j<507 with 1/sqrt(32) scale; one wave per row; bf16 output
// padded with zeros to 512 columns.
__global__ void __launch_bounds__(256)
softmax_rows(const float* __restrict__ scores, unsigned short* __restrict__ prob)
{
    int wave = threadIdx.x >> 5;
    int lane = threadIdx.x & 31;
    long long row = (long long)blockIdx.x * 8 + wave;
    if (row >= (long long)64 * S_LEN) return;
    int bn = (int)(row / S_LEN);
    int i  = (int)(row % S_LEN);
    const float* src      = scores + ((long long)bn * SPAD + i) * SPAD;
    unsigned short* dst   = prob   + ((long long)bn * SPAD + i) * SPAD;
    const float scale = 0.17677669529663687f;   // 1/sqrt(32)

    float vals[16];
    float mx = -3.0e38f;
    #pragma unroll
    for (int t = 0; t < 16; ++t) {
        int j = lane + t * 32;
        float v = (j < S_LEN) ? src[j] * scale : -3.0e38f;
        vals[t] = v;
        mx = fmaxf(mx, v);
    }
    for (int s = 16; s > 0; s >>= 1) mx = fmaxf(mx, __shfl_xor(mx, s, 32));
    float sum = 0.f;
    #pragma unroll
    for (int t = 0; t < 16; ++t) {
        int j = lane + t * 32;
        float e = (j < S_LEN) ? __expf(vals[t] - mx) : 0.f;
        vals[t] = e;
        sum += e;
    }
    for (int s = 16; s > 0; s >>= 1) sum += __shfl_xor(sum, s, 32);
    float inv = 1.f / sum;
    #pragma unroll
    for (int t = 0; t < 16; ++t) {
        int j = lane + t * 32;
        dst[j] = (j < S_LEN) ? f2bf(vals[t] * inv) : (unsigned short)0;
    }
}

// Mean over sequence -> pooled bf16 [16,128]
__global__ void meanpool(const float* __restrict__ wf, unsigned short* __restrict__ pooledb)
{
    int t = blockIdx.x * blockDim.x + threadIdx.x;
    if (t >= BSZ * DMODEL) return;
    int b = t >> 7, c = t & 127;
    float s = 0.f;
    for (int q = 0; q < S_LEN; ++q)
        s += wf[(long long)(q * BSZ + b) * DMODEL + c];
    pooledb[t] = f2bf(s * (1.0f / (float)S_LEN));
}

// ---------------------------------------------------------------------------
extern "C" void kernel_launch(void* const* d_in, const int* in_sizes, int n_in,
                              void* d_out, int out_size, void* d_ws, size_t ws_size,
                              hipStream_t stream)
{
    (void)in_sizes; (void)n_in; (void)out_size; (void)ws_size;
    const float* x      = (const float*)d_in[0];
    const float* conv_w = (const float*)d_in[1];
    const float* conv_b = (const float*)d_in[2];
    const float* qkv_w  = (const float*)d_in[3];
    const float* r_w    = (const float*)d_in[4];
    const float* o_w    = (const float*)d_in[5];
    const float* ff_w1  = (const float*)d_in[6];
    const float* ff_b1  = (const float*)d_in[7];
    const float* ff_w2  = (const float*)d_in[8];
    const float* ff_b2  = (const float*)d_in[9];
    const float* rwb    = (const float*)d_in[10];
    const float* rrb    = (const float*)d_in[11];
    const float* fc_w   = (const float*)d_in[12];
    const float* fc_b   = (const float*)d_in[13];
    float* out = (float*)d_out;

    char* ws = (char*)d_ws;
    size_t off = 0;
    auto alloc = [&](size_t bytes) -> char* {
        char* p = ws + off;
        off = (off + bytes + 255) & ~(size_t)255;
        return p;
    };
    float*          p0f   = (float*)         alloc((size_t)MROWS * DMODEL * 4);
    unsigned short* p0b   = (unsigned short*)alloc((size_t)MROWS * DMODEL * 2);
    float*          p1f   = (float*)         alloc((size_t)MROWS * DMODEL * 4);
    unsigned short* p1b   = (unsigned short*)alloc((size_t)MROWS * DMODEL * 2);
    float*          qkvf  = (float*)         alloc((size_t)MROWS * 384 * 4);
    unsigned short* Qac   = (unsigned short*)alloc((size_t)64 * SPAD * DHEAD * 2);
    unsigned short* Qbd   = (unsigned short*)alloc((size_t)64 * SPAD * DHEAD * 2);
    unsigned short* Kt    = (unsigned short*)alloc((size_t)64 * SPAD * DHEAD * 2);
    unsigned short* Vt    = (unsigned short*)alloc((size_t)64 * DHEAD * SPAD * 2);
    float*          sco   = (float*)         alloc((size_t)64 * SPAD * SPAD * 4);
    unsigned short* prob  = (unsigned short*)alloc((size_t)64 * SPAD * SPAD * 2);
    unsigned short* attnb = (unsigned short*)alloc((size_t)MROWS * DMODEL * 2);
    unsigned short* ffh   = (unsigned short*)alloc((size_t)MROWS * DINNER * 2);
    unsigned short* posb  = (unsigned short*)alloc((size_t)RPAD * DMODEL * 2);
    unsigned short* rkb   = (unsigned short*)alloc((size_t)RPAD * DMODEL * 2);
    unsigned short* qkvwT = (unsigned short*)alloc((size_t)384 * DMODEL * 2);
    unsigned short* rwT   = (unsigned short*)alloc((size_t)DMODEL * DMODEL * 2);
    unsigned short* owT   = (unsigned short*)alloc((size_t)DMODEL * DMODEL * 2);
    unsigned short* f1T   = (unsigned short*)alloc((size_t)DINNER * DMODEL * 2);
    unsigned short* f2T   = (unsigned short*)alloc((size_t)DMODEL * DINNER * 2);
    unsigned short* fcwT  = (unsigned short*)alloc((size_t)256 * DMODEL * 2);
    unsigned short* poolb = (unsigned short*)alloc((size_t)BSZ * DMODEL * 2);

    auto gemm = [&](int NB,
                    const unsigned short* A, int lda, long long aO, long long aI,
                    const unsigned short* Bt, int ldb, long long bO, long long bI,
                    float* of, unsigned short* ob, int ldc, long long cO, long long cI,
                    const float* bias, const float* res,
                    int M, int N, int K, int Mvalid, int relu, int batch) {
        int tiles = (M / 16) * ((N / 16) / NB);
        dim3 grid((tiles + 3) / 4, batch);
        if (NB == 4)
            gemm_wmma<4><<<grid, 128, 0, stream>>>(A, lda, aO, aI, Bt, ldb, bO, bI,
                                                   of, ob, ldc, cO, cI, bias, res,
                                                   M, N, K, Mvalid, relu);
        else
            gemm_wmma<2><<<grid, 128, 0, stream>>>(A, lda, aO, aI, Bt, ldb, bO, bI,
                                                   of, ob, ldc, cO, cI, bias, res,
                                                   M, N, K, Mvalid, relu);
    };
    auto transp = [&](const float* in, unsigned short* o, int R, int C) {
        transpose_f32_bf16<<<(R * C + 255) / 256, 256, 0, stream>>>(in, o, R, C);
    };

    // --- frontend ---
    conv_front<<<(MROWS * DMODEL + 255) / 256, 256, 0, stream>>>(x, conv_w, conv_b, p0f, p0b);
    posemb_kernel<<<(RPAD * DMODEL + 255) / 256, 256, 0, stream>>>(posb);
    transp(fc_w, fcwT, DMODEL, 256);

    for (int L = 0; L < 2; ++L) {
        transp(qkv_w + (size_t)L * DMODEL * 384,  qkvwT, DMODEL, 384);
        transp(r_w   + (size_t)L * DMODEL * DMODEL, rwT, DMODEL, DMODEL);
        transp(o_w   + (size_t)L * DMODEL * DMODEL, owT, DMODEL, DMODEL);
        transp(ff_w1 + (size_t)L * DMODEL * DINNER, f1T, DMODEL, DINNER);
        transp(ff_w2 + (size_t)L * DINNER * DMODEL, f2T, DINNER, DMODEL);

        // qkv projection: [8112,128] x [128,384]
        gemm(4, p0b, 128, 0, 0, qkvwT, 128, 0, 0, qkvf, nullptr, 384, 0, 0,
             nullptr, nullptr, MROWS, 384, DMODEL, MROWS, 0, 1);
        // r_head_k: [1024,128] x [128,128] -> bf16 [1024,128]
        gemm(4, posb, 128, 0, 0, rwT, 128, 0, 0, nullptr, rkb, 128, 0, 0,
             nullptr, nullptr, RPAD, DMODEL, DMODEL, RPAD, 0, 1);

        qkv_prep<<<(64 * SPAD * DHEAD + 255) / 256, 256, 0, stream>>>(
            qkvf, rwb, rrb, Qac, Qbd, Kt, Vt);

        // AC: per (b,n): (q + r_w_bias)[512,32] x K^T -> scores[512,512]
        gemm(4, Qac, 32, 4LL * SPAD * DHEAD, (long long)SPAD * DHEAD,
             Kt,  32, 4LL * SPAD * DHEAD, (long long)SPAD * DHEAD,
             sco, nullptr, SPAD, 4LL * SPAD * SPAD, (long long)SPAD * SPAD,
             nullptr, nullptr, SPAD, SPAD, DHEAD, SPAD, 0, 64);

        // BD with fused rel_shift scatter-add into scores
        {
            dim3 grid(((SPAD / 16) * ((RPAD / 16) / 4) + 3) / 4, 64);
            bd_wmma<<<grid, 128, 0, stream>>>(Qbd, rkb, sco);
        }

        softmax_rows<<<(64 * S_LEN + 7) / 8, 256, 0, stream>>>(sco, prob);

        // attn_vec: prob[512,512] x Vt^T -> attnb [(s*16+b), n*32+d]
        gemm(2, prob, SPAD, 4LL * SPAD * SPAD, (long long)SPAD * SPAD,
             Vt,   SPAD, 4LL * DHEAD * SPAD, (long long)DHEAD * SPAD,
             nullptr, attnb, 2048, 128, 32,
             nullptr, nullptr, SPAD, DHEAD, SPAD, S_LEN, 0, 64);

        // output projection + residual: p1 = p0 + attn @ o_w
        gemm(4, attnb, 128, 0, 0, owT, 128, 0, 0, p1f, p1b, 128, 0, 0,
             nullptr, p0f, MROWS, DMODEL, DMODEL, MROWS, 0, 1);
        // ff1: relu(p1 @ W1 + b1) -> ffh (bf16)
        gemm(4, p1b, 128, 0, 0, f1T, 128, 0, 0, nullptr, ffh, 512, 0, 0,
             ff_b1 + (size_t)L * DINNER, nullptr, MROWS, DINNER, DMODEL, MROWS, 1, 1);
        // ff2 + residual: p0 = p1 + ffh @ W2 + b2
        gemm(4, ffh, 512, 0, 0, f2T, 512, 0, 0, p0f, p0b, 128, 0, 0,
             ff_b2 + (size_t)L * DMODEL, p1f, MROWS, DMODEL, DINNER, MROWS, 0, 1);
    }

    meanpool<<<(BSZ * DMODEL + 255) / 256, 256, 0, stream>>>(p0f, poolb);
    // classifier: [16,128] x [128,256] + fc_b -> d_out (f32)
    gemm(4, poolb, 128, 0, 0, fcwT, 128, 0, 0, out, nullptr, 256, 0, 0,
         fc_b, nullptr, BSZ, 256, DMODEL, BSZ, 0, 1);
}
